// KACacheAttention_33749853012305
// MI455X (gfx1250) — matrix operand order repaired and checked by reading, not compiled, not run on hardware
//
#include <hip/hip_runtime.h>
#include <hip/hip_bf16.h>

// ---------------------------------------------------------------------------
// Problem constants (match reference)
// ---------------------------------------------------------------------------
#define BB      2
#define SS      1024
#define NH      16
#define HD      128
#define HID     2048
#define N3      6144          // 3*HID
#define CACHED  1024
#define TOT     2048          // CACHED + SS
#define ROT     32
#define SCALE   0.08838834764831845f   // 1/sqrt(128)

typedef __attribute__((ext_vector_type(16))) __bf16 v16bf;
typedef __attribute__((ext_vector_type(8)))  __bf16 v8bf;
typedef __attribute__((ext_vector_type(8)))  float  v8f;
typedef int gv4i __attribute__((vector_size(16)));   // matches builtin signature

#define DEV static __device__ __forceinline__

#if __has_builtin(__builtin_amdgcn_global_load_async_to_lds_b128)
#define HAVE_ASYNC_LDS 1
#else
#define HAVE_ASYNC_LDS 0
#endif

DEV void wait_asynccnt0() {
#if __has_builtin(__builtin_amdgcn_s_wait_asynccnt)
  __builtin_amdgcn_s_wait_asynccnt(0);
#else
  asm volatile("s_wait_asynccnt 0" ::: "memory");
#endif
}

#if HAVE_ASYNC_LDS
// async 16B global -> LDS copy (per-lane addresses), tracked by ASYNCcnt
DEV void async_cp16(const __bf16* g, __bf16* l) {
  __builtin_amdgcn_global_load_async_to_lds_b128(
      (__attribute__((address_space(1))) gv4i*)g,
      (__attribute__((address_space(3))) gv4i*)l, 0, 0);
}
#endif

DEV v8bf ld8(const __bf16* p) { return *(const v8bf*)p; }

DEV v16bf cat8(v8bf lo, v8bf hi) {
  v16bf r;
#pragma unroll
  for (int i = 0; i < 8; ++i) { r[i] = lo[i]; r[8 + i] = hi[i]; }
  return r;
}

// A fragment 16x32 bf16: lane row = lane%16; VGPR0-3 hold K = L*8..L*8+7,
// VGPR4-7 hold K = 16+L*8..16+L*8+7 (L = lane/16). Two b128 loads.
DEV v16bf load_afrag(const __bf16* row, int Lh) {
  return cat8(ld8(row + Lh * 8), ld8(row + 16 + Lh * 8));
}
// B fragment 32x16 bf16 from an N-major operand: lane col = lane%16,
// K = L*8..L*8+7 then 16+L*8..16+L*8+7. Two b128 loads.
DEV v16bf load_bfrag(const __bf16* row, int Lh) {
  return cat8(ld8(row + Lh * 8), ld8(row + 16 + Lh * 8));
}

DEV v8f wmma_bf(v16bf a, v16bf b, v8f c) {
  return __builtin_amdgcn_wmma_f32_16x16x32_bf16(
      false, a, false, b, (short)0, c, false, false);
}

DEV float rmax16(float v) {
#pragma unroll
  for (int m = 1; m < 16; m <<= 1) v = fmaxf(v, __shfl_xor(v, m, 16));
  return v;
}
DEV float rsum16(float v) {
#pragma unroll
  for (int m = 1; m < 16; m <<= 1) v += __shfl_xor(v, m, 16);
  return v;
}

// ---------------------------------------------------------------------------
// Pack / transpose / convert kernels (f32 -> bf16 staging in workspace)
// ---------------------------------------------------------------------------
__global__ void k_pack_f32_bf16(const float* __restrict__ src,
                                __bf16* __restrict__ dst, int n) {
  int i = blockIdx.x * blockDim.x + threadIdx.x;
  if (i < n) dst[i] = (__bf16)src[i];
}

// dst[n*K + k] = src[k*N + n]  (weight transpose to N-major bf16)
__global__ void k_pack_T(const float* __restrict__ src,
                         __bf16* __restrict__ dst, int N, int K) {
  int i = blockIdx.x * blockDim.x + threadIdx.x;
  if (i >= N * K) return;
  int n = i / K, k = i - n * K;
  dst[i] = (__bf16)src[(size_t)k * N + n];
}

// cached_k -> fullk_bf[:, :, :1024, :] (bf16) and full_key out first half (f32)
__global__ void k_pack_cachedk(const float* __restrict__ ck,
                               __bf16* __restrict__ fullk,
                               float* __restrict__ fullkey_out) {
  int i = blockIdx.x * blockDim.x + threadIdx.x;
  if (i >= BB * NH * CACHED * HD) return;
  int d = i & 127, t = (i >> 7) & 1023, h = (i >> 17) & 15, b = i >> 21;
  float v = ck[i];
  size_t o = ((size_t)(b * NH + h) * TOT + t) * HD + d;
  fullk[o] = (__bf16)v;
  fullkey_out[o] = v;
}

// cached_a -> fullvT[:, :, d, :1024] (bf16, transposed) and new_cached_a first half
__global__ void k_pack_cacheda(const float* __restrict__ ca,
                               __bf16* __restrict__ fullvT,
                               float* __restrict__ nca_out) {
  int i = blockIdx.x * blockDim.x + threadIdx.x;
  if (i >= BB * NH * CACHED * HD) return;
  int d = i & 127, t = (i >> 7) & 1023, h = (i >> 17) & 15, b = i >> 21;
  float v = ca[i];
  fullvT[((size_t)(b * NH + h) * HD + d) * TOT + t] = (__bf16)v;
  nca_out[((size_t)(b * NH + h) * TOT + t) * HD + d] = v;
}

// ---------------------------------------------------------------------------
// 64x32 macro-tile bf16 GEMM core, K = HID (compile-time): 8 WMMAs per
// 32-wide K step; 8 accumulators (~64 VGPRs) + 6 fragments -> no spills.
// Row offsets mi*16*HID fold into the load-instruction immediate.
// ---------------------------------------------------------------------------
DEV void tile_gemm64x32(const __bf16* __restrict__ A, const __bf16* __restrict__ BT,
                        int m0, int n0, v8f c[4][2]) {
  int lane = threadIdx.x & 31;
  int col = lane & 15, Lh = lane >> 4;
  const __bf16* abase = A + (size_t)(m0 + col) * HID;
  const __bf16* bbase = BT + (size_t)(n0 + col) * HID;
#pragma unroll
  for (int mi = 0; mi < 4; ++mi)
#pragma unroll
    for (int ni = 0; ni < 2; ++ni) c[mi][ni] = (v8f)0.0f;

  for (int k0 = 0; k0 < HID; k0 += 32) {
    v16bf a[4], b[2];
#pragma unroll
    for (int mi = 0; mi < 4; ++mi)
      a[mi] = load_afrag(abase + mi * (16 * HID) + k0, Lh);
#pragma unroll
    for (int ni = 0; ni < 2; ++ni)
      b[ni] = load_bfrag(bbase + ni * (16 * HID) + k0, Lh);
#pragma unroll
    for (int mi = 0; mi < 4; ++mi)
#pragma unroll
      for (int ni = 0; ni < 2; ++ni)
        c[mi][ni] = wmma_bf(a[mi], b[ni], c[mi][ni]);
  }
}

// QKV GEMM: [2048,2048] x [2048,6144] + bias, scattered to q/k/v raw (bf16)
__global__ void k_gemm_qkv(const __bf16* __restrict__ hsb,
                           const __bf16* __restrict__ WqkvT,
                           const float* __restrict__ bqkv,
                           __bf16* __restrict__ qraw,
                           __bf16* __restrict__ kraw,
                           __bf16* __restrict__ vraw) {
  int wave = blockIdx.x * (blockDim.x >> 5) + (threadIdx.x >> 5);
  int nt = wave % (N3 / 32);
  int mt = wave / (N3 / 32);
  v8f c[4][2];
  tile_gemm64x32(hsb, WqkvT, mt * 64, nt * 32, c);
  int lane = threadIdx.x & 31, col = lane & 15, Lh = lane >> 4;
#pragma unroll
  for (int ni = 0; ni < 2; ++ni) {
    int n = nt * 32 + ni * 16 + col;
    int which = n >> 11, h = (n >> 7) & 15, d = n & 127;
    __bf16* base = (which == 0) ? qraw : ((which == 1) ? kraw : vraw);
    float bias = bqkv[n];
#pragma unroll
    for (int mi = 0; mi < 4; ++mi) {
#pragma unroll
      for (int r = 0; r < 8; ++r) {
        int m = mt * 64 + mi * 16 + r + 8 * Lh;
        int b = m >> 10, s = m & 1023;
        base[((size_t)(b * NH + h) * SS + s) * HD + d] = (__bf16)(c[mi][ni][r] + bias);
      }
    }
  }
}

// Dense GEMM: out = attn_flat @ W_dense + b_dense  (f32 out)
__global__ void k_gemm_dense(const __bf16* __restrict__ Ab,
                             const __bf16* __restrict__ WdT,
                             const float* __restrict__ bd,
                             float* __restrict__ out) {
  int wave = blockIdx.x * (blockDim.x >> 5) + (threadIdx.x >> 5);
  int nt = wave % (HID / 32);
  int mt = wave / (HID / 32);
  v8f c[4][2];
  tile_gemm64x32(Ab, WdT, mt * 64, nt * 32, c);
  int lane = threadIdx.x & 31, col = lane & 15, Lh = lane >> 4;
#pragma unroll
  for (int ni = 0; ni < 2; ++ni) {
    int n = nt * 32 + ni * 16 + col;
    float bias = bd[n];
#pragma unroll
    for (int mi = 0; mi < 4; ++mi) {
#pragma unroll
      for (int r = 0; r < 8; ++r) {
        int m = mt * 64 + mi * 16 + r + 8 * Lh;
        out[(size_t)m * HID + n] = c[mi][ni][r] + bias;
      }
    }
  }
}

// ---------------------------------------------------------------------------
// RoPE + cache assembly: q/k get rotary on first 32 dims; write roped q (bf16),
// fullk second half (bf16 + f32 out), and V transposed into fullvT.
// ---------------------------------------------------------------------------
__global__ void k_rope(const __bf16* __restrict__ qraw,
                       const __bf16* __restrict__ kraw,
                       const __bf16* __restrict__ vraw,
                       __bf16* __restrict__ qbf,
                       __bf16* __restrict__ fullk,
                       __bf16* __restrict__ fullvT,
                       float* __restrict__ fullkey_out) {
  int i = blockIdx.x * blockDim.x + threadIdx.x;
  if (i >= BB * NH * SS * HD) return;
  int d = i & 127, s = (i >> 7) & 1023, h = (i >> 17) & 15, b = i >> 21;
  size_t bh = (size_t)(b * NH + h);
  size_t src = (bh * SS + s) * HD + d;

  float q = (float)qraw[src];
  float k = (float)kraw[src];
  if (d < ROT) {
    int j = d & 15;
    float pos = (float)(CACHED + s);
    float ang = pos * __expf(-(float)(2 * j) * (1.0f / 32.0f) * 9.210340371976184f);
    float cs = __cosf(ang), sn = __sinf(ang);
    int pidx = (d < 16) ? (d + 16) : (d - 16);
    float sgn = (d < 16) ? -1.0f : 1.0f;
    float qp = (float)qraw[(bh * SS + s) * HD + pidx];
    float kp = (float)kraw[(bh * SS + s) * HD + pidx];
    q = q * cs + sgn * qp * sn;
    k = k * cs + sgn * kp * sn;
  }
  qbf[src] = (__bf16)q;
  size_t ko = (bh * TOT + CACHED + s) * HD + d;
  fullk[ko] = (__bf16)k;
  fullkey_out[ko] = k;
  fullvT[(bh * HD + d) * TOT + CACHED + s] = vraw[src];
}

// ---------------------------------------------------------------------------
// Flash attention: one wave (block of 32) per (b, h, 16-row q tile).
// Q(16x128) x K^T streaming in 32-wide KV steps, online softmax, P x V.
// Q tile is staged through LDS via gfx1250 async global->LDS loads.
// ---------------------------------------------------------------------------
__global__ void __launch_bounds__(32)
k_attn(const __bf16* __restrict__ qbf,
       const __bf16* __restrict__ fullk,
       const __bf16* __restrict__ fullvT,
       float* __restrict__ nca_out,          // new_cached_a (writes 2nd half)
       __bf16* __restrict__ attnbf) {        // [B,S,H*D] for dense GEMM
  __shared__ __align__(16) __bf16 pls[16 * 32];
  __shared__ __align__(16) __bf16 qls[16 * HD];

  int tile = blockIdx.x;
  int qt = tile & 63;
  int h = (tile >> 6) & 15;
  int b = tile >> 10;
  int qbase = qt * 16;
  int lane = threadIdx.x & 31, col = lane & 15, Lh = lane >> 4;
  size_t bh = (size_t)(b * NH + h);

  // ---- stage the contiguous 16x128 bf16 Q tile into LDS ----
  const __bf16* qtile = qbf + (bh * SS + qbase) * HD;   // 4 KB contiguous
#if HAVE_ASYNC_LDS
  // 256 16-byte chunks, 32 lanes per issue -> 8 async b128 issues
#pragma unroll
  for (int it = 0; it < 8; ++it) {
    int idx = it * 32 + lane;
    async_cp16(qtile + idx * 8, qls + idx * 8);
  }
  wait_asynccnt0();
  __syncthreads();
#else
#pragma unroll
  for (int it = 0; it < 8; ++it) {
    int idx = it * 32 + lane;
    *(v8bf*)(qls + idx * 8) = ld8(qtile + idx * 8);
  }
  __syncthreads();
#endif

  // Load Q fragments (4 chunks of K=32 over head dim 128) from LDS.
  v16bf qf[4];
#pragma unroll
  for (int kk = 0; kk < 4; ++kk)
    qf[kk] = load_afrag(qls + col * HD + kk * 32, Lh);

  float M[8], Ls[8];
  v8f O[8];
#pragma unroll
  for (int r = 0; r < 8; ++r) { M[r] = -1e30f; Ls[r] = 0.0f; O[r] = (v8f)0.0f; }

  // Causal bound: row i=qbase+m allows j <= CACHED + i.
  int kv_end = ((CACHED + qbase + 16) + 31) & ~31;
  if (kv_end > TOT) kv_end = TOT;

  for (int kv0 = 0; kv0 < kv_end; kv0 += 32) {
    // ---- scores S (16 x 32) = Q x K^T, two 16x16 column blocks ----
    v8f s0 = (v8f)0.0f, s1 = (v8f)0.0f;
    const __bf16* k0row = fullk + (bh * TOT + kv0 + col) * HD;
    const __bf16* k1row = fullk + (bh * TOT + kv0 + 16 + col) * HD;
#pragma unroll
    for (int kk = 0; kk < 4; ++kk) {
      s0 = wmma_bf(qf[kk], load_bfrag(k0row + kk * 32, Lh), s0);
      s1 = wmma_bf(qf[kk], load_bfrag(k1row + kk * 32, Lh), s1);
    }

    // ---- scale + causal mask + online softmax ----
#pragma unroll
    for (int r = 0; r < 8; ++r) {
      int i = qbase + r + 8 * Lh;
      int j0 = kv0 + col, j1 = kv0 + 16 + col;
      float a0 = (j0 > CACHED + i) ? -1e30f : s0[r] * SCALE;
      float a1 = (j1 > CACHED + i) ? -1e30f : s1[r] * SCALE;
      float mnew = fmaxf(M[r], rmax16(fmaxf(a0, a1)));
      float alpha = __expf(M[r] - mnew);
      float p0 = __expf(a0 - mnew);
      float p1 = __expf(a1 - mnew);
      Ls[r] = Ls[r] * alpha + rsum16(p0 + p1);
      M[r] = mnew;
      int m = r + 8 * Lh;
      pls[m * 32 + col] = (__bf16)p0;
      pls[m * 32 + 16 + col] = (__bf16)p1;
      // rescale running output row
#pragma unroll
      for (int t = 0; t < 8; ++t) O[t][r] *= alpha;
    }
    __syncthreads();

    // ---- P (16x32) x V (32x128): P from LDS as A-fragment ----
    v16bf pf = load_afrag(pls + col * 32, Lh);
#pragma unroll
    for (int t = 0; t < 8; ++t) {
      const __bf16* vrow = fullvT + (bh * HD + t * 16 + col) * TOT + kv0;
      O[t] = wmma_bf(pf, load_bfrag(vrow, Lh), O[t]);
    }
    __syncthreads();
  }

  // ---- normalize and write: new_cached_a (f32) + attn bf16 for dense GEMM ----
#pragma unroll
  for (int r = 0; r < 8; ++r) {
    int m = r + 8 * Lh;
    int srow = qbase + m;
    float inv = 1.0f / Ls[r];
#pragma unroll
    for (int t = 0; t < 8; ++t) {
      int d = t * 16 + col;
      float val = O[t][r] * inv;
      nca_out[(bh * TOT + CACHED + srow) * HD + d] = val;
      attnbf[((size_t)(b * SS + srow)) * HID + h * HD + d] = (__bf16)val;
    }
  }
}

// ---------------------------------------------------------------------------
// Host launcher
// ---------------------------------------------------------------------------
extern "C" void kernel_launch(void* const* d_in, const int* in_sizes, int n_in,
                              void* d_out, int out_size, void* d_ws, size_t ws_size,
                              hipStream_t stream) {
  const float* hs   = (const float*)d_in[0];
  const float* ck   = (const float*)d_in[2];
  const float* ca   = (const float*)d_in[3];
  const float* Wqkv = (const float*)d_in[4];
  const float* bqkv = (const float*)d_in[5];
  const float* Wd   = (const float*)d_in[6];
  const float* bd   = (const float*)d_in[7];

  float* out_dense   = (float*)d_out;                               // [B,S,HID]
  float* out_fullkey = out_dense + (size_t)BB * SS * HID;           // [B,H,TOT,HD]
  float* out_nca     = out_fullkey + (size_t)BB * NH * TOT * HD;    // [B,H,TOT,HD]

  // ---- workspace carve-up (bf16 staging, ~115 MB) ----
  __bf16* w = (__bf16*)d_ws;
  __bf16* hsb    = w;                       w += (size_t)HID * HID;        // 2048x2048
  __bf16* WqkvT  = w;                       w += (size_t)N3 * HID;         // 6144x2048
  __bf16* WdT    = w;                       w += (size_t)HID * HID;
  __bf16* qraw   = w;                       w += (size_t)BB * NH * SS * HD;
  __bf16* kraw   = w;                       w += (size_t)BB * NH * SS * HD;
  __bf16* vraw   = w;                       w += (size_t)BB * NH * SS * HD;
  __bf16* qbf    = w;                       w += (size_t)BB * NH * SS * HD;
  __bf16* fullk  = w;                       w += (size_t)BB * NH * TOT * HD;
  __bf16* fullvT = w;                       w += (size_t)BB * NH * TOT * HD;
  __bf16* attnbf = w;                       w += (size_t)BB * SS * HID;

  const int TPB = 256;
  // pack/convert
  k_pack_f32_bf16<<<(HID * HID + TPB - 1) / TPB, TPB, 0, stream>>>(hs, hsb, HID * HID);
  k_pack_T<<<(N3 * HID + TPB - 1) / TPB, TPB, 0, stream>>>(Wqkv, WqkvT, N3, HID);
  k_pack_T<<<(HID * HID + TPB - 1) / TPB, TPB, 0, stream>>>(Wd, WdT, HID, HID);
  {
    int n = BB * NH * CACHED * HD;
    k_pack_cachedk<<<(n + TPB - 1) / TPB, TPB, 0, stream>>>(ck, fullk, out_fullkey);
    k_pack_cacheda<<<(n + TPB - 1) / TPB, TPB, 0, stream>>>(ca, fullvT, out_nca);
  }
  // QKV GEMM: (2048/64)*(6144/32) = 32*192 macro-tiles, 8 waves/block
  k_gemm_qkv<<<(32 * 192) / 8, TPB, 0, stream>>>(hsb, WqkvT, bqkv, qraw, kraw, vraw);
  // RoPE + cache assembly
  k_rope<<<(BB * NH * SS * HD + TPB - 1) / TPB, TPB, 0, stream>>>(
      qraw, kraw, vraw, qbf, fullk, fullvT, out_fullkey);
  // Flash attention: one wave per (b,h,qtile)
  k_attn<<<BB * NH * (SS / 16), 32, 0, stream>>>(qbf, fullk, fullvT, out_nca, attnbf);
  // Dense output GEMM: (2048/64)*(2048/32) = 32*64 macro-tiles, 8 waves/block
  k_gemm_dense<<<(32 * 64) / 8, TPB, 0, stream>>>(attnbf, WdT, bd, out_dense);
}
